// L1KnowledgeMoE_58274116272205
// MI455X (gfx1250) — matrix operand order repaired and checked by reading, not compiled, not run on hardware
//
#include <hip/hip_runtime.h>
#include <stdint.h>

// ---------------------------------------------------------------------------
// MoE (top-2, 64 experts) + LayerNorm for MI455X (gfx1250, wave32, WMMA).
// Sparse dispatch: router -> per-expert token buckets -> bf16 WMMA expert
// GEMMs (fp32 accumulate) -> per-slot contribution rows -> LN combine.
// Memory-bound on one fp32 pass over W1/W2 (402 MB @ 23.3 TB/s); weights are
// converted to bf16 in-register while staging to LDS (no extra HBM traffic).
// ---------------------------------------------------------------------------

#define D_MODEL      1536
#define NUM_EXPERTS  64
#define D_HIDDEN     512
#define T_TOKENS     4096
#define LN_EPS       1e-5f

#define TM        64      // token tile (M) per expert workgroup
#define KC        32      // WMMA K per step (bf16 16x16x32)
#define THREADS2  512     // 16 waves

typedef __attribute__((ext_vector_type(16))) __bf16 v16bf;
typedef __attribute__((ext_vector_type(8)))  float  v8f;

// Hardware f32->bf16 convert (RNE); compiler picks v_cvt / packed forms.
__device__ __forceinline__ uint16_t f2bf(float f) {
  __bf16 h = (__bf16)f;
  return __builtin_bit_cast(uint16_t, h);
}
__device__ __forceinline__ uint32_t pack2bf(float lo, float hi) {
  union { __bf16 h[2]; uint32_t u; } r;
  r.h[0] = (__bf16)lo; r.h[1] = (__bf16)hi;
  return r.u;
}

// A-matrix fragment (16x32 bf16), per documented CDNA5 layout:
// lane -> M = lane%16 ; VGPR j: K = (j>=4?16:0) + (lane/16)*8 + (j&3)*2 (pair)
__device__ __forceinline__ v16bf load_afrag(const uint16_t* rowbase, int lane) {
  int kh8 = (lane >> 4) * 8;
  union { uint32_t u[8]; v16bf v; } r;
  const uint32_t* p = (const uint32_t*)(rowbase + kh8);
  const uint32_t* q = (const uint32_t*)(rowbase + 16 + kh8);
#pragma unroll
  for (int i = 0; i < 4; ++i) { r.u[i] = p[i]; r.u[4 + i] = q[i]; }
  return r.v;
}

// B-matrix fragment (32x16 bf16): lane -> N = lane%16 ; element s: K same as A
// (table order), gathered down a column of the K-major [k][n] LDS tile.
__device__ __forceinline__ v16bf load_bfrag(const uint16_t* p16, int ld, int lane) {
  int kh8 = (lane >> 4) * 8;
  union { uint16_t h[16]; v16bf v; } r;
#pragma unroll
  for (int i = 0; i < 8; ++i) {
    r.h[i] = p16[(kh8 + i) * ld];
    r.h[8 + i] = p16[(16 + kh8 + i) * ld];
  }
  return r.v;
}

__device__ __forceinline__ v8f v8f_zero() {
  v8f z = {0.f, 0.f, 0.f, 0.f, 0.f, 0.f, 0.f, 0.f};
  return z;
}

// ---------------------------------------------------------------------------
__global__ __launch_bounds__(64) void moe_zero(int* __restrict__ counts) {
  if (threadIdx.x < NUM_EXPERTS) counts[threadIdx.x] = 0;
}

// One block per token: logits, top-2, softmax(2), bucket append, bf16(x) copy.
__global__ __launch_bounds__(64) void moe_router(
    const float* __restrict__ x, const float* __restrict__ Wr,
    uint16_t* __restrict__ xb, int* __restrict__ counts,
    int* __restrict__ ents, float* __restrict__ bw) {
  __shared__ float xs[D_MODEL];
  __shared__ float lg[NUM_EXPERTS];
  const int t = blockIdx.x;
  const int tid = threadIdx.x;
  const float* xr = x + (size_t)t * D_MODEL;
  for (int d = tid * 2; d < D_MODEL; d += 128) {
    float v0 = xr[d], v1 = xr[d + 1];
    xs[d] = v0; xs[d + 1] = v1;
    *(uint32_t*)(xb + (size_t)t * D_MODEL + d) = pack2bf(v0, v1);
  }
  __syncthreads();
  const float* wr = Wr + (size_t)tid * D_MODEL;
  float acc = 0.f;
  for (int d = 0; d < D_MODEL; d += 4) {
    acc += xs[d] * wr[d] + xs[d + 1] * wr[d + 1] +
           xs[d + 2] * wr[d + 2] + xs[d + 3] * wr[d + 3];
  }
  lg[tid] = acc;
  __syncthreads();
  if (tid == 0) {
    int i0 = 0; float b0 = lg[0];
    for (int e = 1; e < NUM_EXPERTS; ++e)
      if (lg[e] > b0) { b0 = lg[e]; i0 = e; }
    int i1 = (i0 == 0) ? 1 : 0; float b1 = lg[i1];
    for (int e = 0; e < NUM_EXPERTS; ++e)
      if (e != i0 && lg[e] > b1) { b1 = lg[e]; i1 = e; }
    float e1 = __expf(b1 - b0);          // b0 >= b1 -> stable
    float w0 = 1.f / (1.f + e1);
    float w1 = e1 * w0;
    int p0 = atomicAdd(&counts[i0], 1);
    ents[i0 * T_TOKENS + p0] = t * 2 + 0;
    bw  [i0 * T_TOKENS + p0] = w0;
    int p1 = atomicAdd(&counts[i1], 1);
    ents[i1 * T_TOKENS + p1] = t * 2 + 1;
    bw  [i1 * T_TOKENS + p1] = w1;
  }
}

// ---------------------------------------------------------------------------
// Expert GEMMs: grid (expert, token-tile). 512 thr = 16 waves.
// Phase1: H(64x512) = silu(X·W1^T)·w via bf16 WMMA, H -> LDS bf16.
// Phase2: out(64x1536) = H·W2^T, scattered to contrib[token_slot].
__global__ __launch_bounds__(THREADS2) void moe_expert(
    const uint16_t* __restrict__ xb,
    const float* __restrict__ W1, const float* __restrict__ W2,
    const int* __restrict__ counts, const int* __restrict__ ents,
    const float* __restrict__ bw, float* __restrict__ contrib) {
  __shared__ uint16_t Abuf[TM * KC];        // 4 KB   X tile, row-major [m][k]
  __shared__ uint16_t Bbuf[KC * 512];       // 32 KB  K-major [k][n]
  __shared__ uint16_t Hbuf[TM * D_HIDDEN];  // 64 KB  H tile, row-major [m][h]
  __shared__ int   tslot_s[TM];
  __shared__ float twt_s[TM];

  const int e   = blockIdx.x;
  const int cnt = counts[e];
  const int t0  = blockIdx.y * TM;
  if (t0 >= cnt) return;
  const int nTok = min(TM, cnt - t0);

  const int tid    = threadIdx.x;
  const int lane   = tid & 31;
  const int wave   = tid >> 5;        // 0..15
  const int m16    = wave & 3;        // 16-row M tile
  const int nq     = wave >> 2;       // N quarter
  const int khalfm = 8 * (lane >> 4); // C/D row offset for this lane half
  const int ncl    = lane & 15;       // C/D column within tile

  if (tid < TM) {
    if (tid < nTok) {
      tslot_s[tid] = ents[e * T_TOKENS + t0 + tid];
      twt_s[tid]   = bw  [e * T_TOKENS + t0 + tid];
    } else {
      tslot_s[tid] = 0;
      twt_s[tid]   = 0.f;
    }
  }

  const float* W1e = W1 + (size_t)e * D_HIDDEN * D_MODEL; // [H, D]
  const float* W2e = W2 + (size_t)e * D_MODEL * D_HIDDEN; // [D, H]

  // ---------------- Phase 1: X(64x1536) @ W1^T -> H(64x512) ----------------
  v8f acc1[8];
#pragma unroll
  for (int i = 0; i < 8; ++i) acc1[i] = v8f_zero();

  // Staging split: thread owns n-pair (n0, n0+1) and a 16-wide K half, so two
  // bf16 converts pack into one b32 LDS store (K-major [k][n] layout).
  const int p1n0 = (tid & 255) * 2;       // 0..510, even
  const int p1kh = (tid >> 8) * 16;       // 0 or 16

  for (int kc = 0; kc < D_MODEL; kc += KC) {
    __syncthreads();
    if (tid < TM) {                      // stage A: 32 bf16 per token row
      const int m = tid;
      uint4 d0 = {0,0,0,0}, d1 = {0,0,0,0}, d2 = {0,0,0,0}, d3 = {0,0,0,0};
      if (m < nTok) {
        const int tok = tslot_s[m] >> 1;
        const uint4* p = (const uint4*)(xb + (size_t)tok * D_MODEL + kc);
        d0 = p[0]; d1 = p[1]; d2 = p[2]; d3 = p[3];
      }
      uint4* q = (uint4*)(Abuf + m * KC);
      q[0] = d0; q[1] = d1; q[2] = d2; q[3] = d3;
    }
    {                                    // stage B: W1[n][kc+kh..+16) -> [k][n]
      const float* ra = W1e + (size_t)p1n0 * D_MODEL + kc + p1kh;
      const float* rb = ra + D_MODEL;
      if (kc + KC < D_MODEL) {
        __builtin_prefetch(ra + KC, 0, 1);
        __builtin_prefetch(rb + KC, 0, 1);
      }
      const float4* a4 = (const float4*)ra;
      const float4* b4 = (const float4*)rb;
#pragma unroll
      for (int i = 0; i < 4; ++i) {
        float4 fa = a4[i];
        float4 fb = b4[i];
        int k = p1kh + i * 4;
        uint32_t* dst = (uint32_t*)(Bbuf + (size_t)k * 512 + p1n0);
        dst[0]         = pack2bf(fa.x, fb.x);   // row k
        dst[256]       = pack2bf(fa.y, fb.y);   // row k+1 (stride 512 elems)
        dst[512]       = pack2bf(fa.z, fb.z);   // row k+2
        dst[768]       = pack2bf(fa.w, fb.w);   // row k+3
      }
    }
    __syncthreads();
    v16bf a = load_afrag(Abuf + (m16 * 16 + ncl) * KC, lane);
    // Software-pipeline B fragments one tile ahead so LDS reads overlap WMMA.
    const uint16_t* bbase = Bbuf + nq * 128 + ncl;
    v16bf bcur = load_bfrag(bbase, 512, lane);
#pragma unroll
    for (int ntile = 0; ntile < 8; ++ntile) {
      v16bf bnxt = bcur;
      if (ntile + 1 < 8) bnxt = load_bfrag(bbase + (ntile + 1) * 16, 512, lane);
      acc1[ntile] = __builtin_amdgcn_wmma_f32_16x16x32_bf16(
          false, a, false, bcur, (short)0, acc1[ntile], false, false);
      bcur = bnxt;
    }
  }

  // Epilogue: silu + routing weight, store bf16 H tile to LDS.
#pragma unroll
  for (int ntile = 0; ntile < 8; ++ntile) {
    int n = nq * 128 + ntile * 16 + ncl;
#pragma unroll
    for (int j = 0; j < 8; ++j) {
      int m = m16 * 16 + khalfm + j;
      float v = acc1[ntile][j];
      float s = v * (1.f / (1.f + __expf(-v)));
      Hbuf[m * D_HIDDEN + n] = f2bf(s * twt_s[m]);
    }
  }

  // ---------------- Phase 2: H(64x512) @ W2^T -> out(64x1536) --------------
  const int p2n0 = (tid & 127) * 2;       // 0..254, even
  const int p2kh = (tid >> 7) * 8;        // 0, 8, 16, 24

  for (int nc = 0; nc < D_MODEL; nc += 256) {
    v8f acc2[4];
#pragma unroll
    for (int i = 0; i < 4; ++i) acc2[i] = v8f_zero();

    for (int kc = 0; kc < D_HIDDEN; kc += KC) {
      __syncthreads();                   // Bbuf reuse fence (+ Hbuf ready)
      {                                  // stage B: W2[nc+n][kc+kh..+8) -> [k][n]
        const float* ra = W2e + (size_t)(nc + p2n0) * D_HIDDEN + kc + p2kh;
        const float* rb = ra + D_HIDDEN;
        if (kc + KC < D_HIDDEN) {
          __builtin_prefetch(ra + KC, 0, 1);
          __builtin_prefetch(rb + KC, 0, 1);
        }
        const float4* a4 = (const float4*)ra;
        const float4* b4 = (const float4*)rb;
#pragma unroll
        for (int i = 0; i < 2; ++i) {
          float4 fa = a4[i];
          float4 fb = b4[i];
          int k = p2kh + i * 4;
          uint32_t* dst = (uint32_t*)(Bbuf + (size_t)k * 256 + p2n0);
          dst[0]   = pack2bf(fa.x, fb.x);       // row k
          dst[128] = pack2bf(fa.y, fb.y);       // row k+1 (stride 256 elems)
          dst[256] = pack2bf(fa.z, fb.z);       // row k+2
          dst[384] = pack2bf(fa.w, fb.w);       // row k+3
        }
      }
      __syncthreads();
      v16bf a = load_afrag(Hbuf + (m16 * 16 + ncl) * D_HIDDEN + kc, lane);
      const uint16_t* bbase = Bbuf + nq * 64 + ncl;
      v16bf bcur = load_bfrag(bbase, 256, lane);
#pragma unroll
      for (int ntile = 0; ntile < 4; ++ntile) {
        v16bf bnxt = bcur;
        if (ntile + 1 < 4) bnxt = load_bfrag(bbase + (ntile + 1) * 16, 256, lane);
        acc2[ntile] = __builtin_amdgcn_wmma_f32_16x16x32_bf16(
            false, a, false, bcur, (short)0, acc2[ntile], false, false);
        bcur = bnxt;
      }
    }
    // Scatter to per-(token,slot) contribution rows (each row written once).
#pragma unroll
    for (int ntile = 0; ntile < 4; ++ntile) {
      int col = nc + nq * 64 + ntile * 16 + ncl;
#pragma unroll
      for (int j = 0; j < 8; ++j) {
        int m = m16 * 16 + khalfm + j;
        if (m < nTok)
          contrib[(size_t)tslot_s[m] * D_MODEL + col] = acc2[ntile][j];
      }
    }
  }
}

// ---------------------------------------------------------------------------
// Combine the two expert contributions per token + LayerNorm.
__global__ __launch_bounds__(256) void moe_ln(
    const float* __restrict__ contrib, const float* __restrict__ gamma,
    const float* __restrict__ beta, float* __restrict__ out) {
  __shared__ float row[D_MODEL];
  __shared__ float ps[8], pss[8];
  const int t = blockIdx.x, tid = threadIdx.x;
  const float* c0 = contrib + (size_t)(2 * t) * D_MODEL;
  const float* c1 = c0 + D_MODEL;
  float s = 0.f, ss = 0.f;
  for (int d = tid; d < D_MODEL; d += 256) {
    float v = c0[d] + c1[d];
    row[d] = v;
    s += v; ss += v * v;
  }
#pragma unroll
  for (int o = 16; o > 0; o >>= 1) {
    s  += __shfl_down(s,  o, 32);
    ss += __shfl_down(ss, o, 32);
  }
  const int lane = tid & 31, wave = tid >> 5;
  if (lane == 0) { ps[wave] = s; pss[wave] = ss; }
  __syncthreads();
  if (tid == 0) {
    float a = 0.f, b = 0.f;
    for (int i = 0; i < 8; ++i) { a += ps[i]; b += pss[i]; }
    ps[0] = a; pss[0] = b;
  }
  __syncthreads();
  const float mu  = ps[0]  * (1.f / D_MODEL);
  const float var = pss[0] * (1.f / D_MODEL) - mu * mu;
  const float inv = rsqrtf(var + LN_EPS);
  for (int d = tid; d < D_MODEL; d += 256)
    out[(size_t)t * D_MODEL + d] = (row[d] - mu) * inv * gamma[d] + beta[d];
}

// ---------------------------------------------------------------------------
extern "C" void kernel_launch(void* const* d_in, const int* in_sizes, int n_in,
                              void* d_out, int out_size, void* d_ws,
                              size_t ws_size, hipStream_t stream) {
  (void)in_sizes; (void)n_in; (void)out_size; (void)ws_size;
  const float* x     = (const float*)d_in[0];
  const float* Wr    = (const float*)d_in[1];
  const float* W1    = (const float*)d_in[2];
  const float* W2    = (const float*)d_in[3];
  const float* gamma = (const float*)d_in[4];
  const float* beta  = (const float*)d_in[5];
  float* out = (float*)d_out;

  char* ws = (char*)d_ws;
  size_t off = 0;
  uint16_t* xb = (uint16_t*)(ws + off);
  off += (size_t)T_TOKENS * D_MODEL * sizeof(uint16_t);            // 12.6 MB
  int* counts = (int*)(ws + off);  off += 1024;
  int* ents   = (int*)(ws + off);  off += (size_t)NUM_EXPERTS * T_TOKENS * 4;
  float* bw   = (float*)(ws + off); off += (size_t)NUM_EXPERTS * T_TOKENS * 4;
  float* contrib = (float*)(ws + off);                              // 50.3 MB
  off += (size_t)T_TOKENS * 2 * D_MODEL * sizeof(float);

  moe_zero<<<1, 64, 0, stream>>>(counts);
  moe_router<<<T_TOKENS, 64, 0, stream>>>(x, Wr, xb, counts, ents, bw);
  dim3 g2(NUM_EXPERTS, T_TOKENS / TM);
  moe_expert<<<g2, THREADS2, 0, stream>>>(xb, W1, W2, counts, ents, bw,
                                          contrib);
  moe_ln<<<T_TOKENS, 256, 0, stream>>>(contrib, gamma, beta, out);
}
